// DeformableTransformerEncoderLayer_38955353375273
// MI455X (gfx1250) — matrix operand order repaired and checked by reading, not compiled, not run on hardware
//
#include <hip/hip_runtime.h>

// ---------------------------------------------------------------------------
// Deformable transformer encoder layer for MI455X (gfx1250, wave32, WMMA).
// GEMMs: v_wmma_f32_16x16x32_f16, double-buffered LDS staging with
// global_load_async_to_lds_b128 (ASYNCcnt) for the activation tiles.
// ---------------------------------------------------------------------------

typedef __attribute__((ext_vector_type(16))) _Float16 v16h;
typedef __attribute__((ext_vector_type(8)))  _Float16 v8h;
typedef __attribute__((ext_vector_type(8)))  float    v8f;

#define NQ_TOTAL 43520   // B * 21760
#define QSUM     21760   // sum of H*W over levels
#define CCH      256
#define FFN      2048

// ---------------------------------------------------------------------------
// zf = feat + pos, relaid out to [b*QSUM + lvl_off + p, 256]
// ---------------------------------------------------------------------------
__global__ void addpos_kernel(const float4* __restrict__ f,
                              const float4* __restrict__ p,
                              float* __restrict__ zf,
                              int HW, int lvl_off) {
  int idx = blockIdx.x * blockDim.x + threadIdx.x;   // over B*HW*64 float4s
  int b   = idx / (HW * 64);
  int rem = idx - b * (HW * 64);
  int pos = rem >> 6;
  int c4  = rem & 63;
  float4 a = f[idx];
  float4 q = p[idx];
  size_t row = (size_t)b * QSUM + lvl_off + pos;
  float4* dst = (float4*)(zf + row * CCH) + c4;
  *dst = make_float4(a.x + q.x, a.y + q.y, a.z + q.z, a.w + q.w);
}

// ---------------------------------------------------------------------------
// Tiled WMMA GEMM:  Out[Nrows x Ncols] = act(A[Nrows x K] * W[K x Ncols] + b)
// Block = 256 threads (8 wave32), tile 128x64, K-step 32, double-buffered.
// A tile staged f32 via async global->LDS copy (no VGPR round trip), converted
// to f16 at fragment build; B tile converted+transposed through VGPRs.
// ---------------------------------------------------------------------------
__global__ __launch_bounds__(256)
void gemm_wmma_kernel(const float* __restrict__ A, const float* __restrict__ W,
                      const float* __restrict__ bias, float* __restrict__ Out,
                      int Nrows, int Kdim, int Ncols, int relu) {
  __shared__ float    lAf[2][128][32];   // 2 x 16 KB, async-staged f32
  __shared__ _Float16 lB[2][64][40];     // [col][k] f16, padded stride

  const int rowBase = blockIdx.y * 128;
  const int colBase = blockIdx.x * 64;
  const int tid     = threadIdx.x;
  const int wave    = tid >> 5;
  const int lane    = tid & 31;
  const int laneLo  = lane & 15;
  const int laneHi  = lane >> 4;
  const int wr      = wave & 3;    // row group: 32 rows each
  const int wc      = wave >> 2;   // col group: 32 cols each

  // ---- async stage of A tile: 128x32 f32 = 1024 b128 copies over 256 thr ----
  auto stageA = [&](int buf, int kb) {
#pragma unroll
    for (int j = 0; j < 4; ++j) {
      int i   = tid + 256 * j;
      int row = i >> 3;
      int c4  = (i & 7) * 4;
      const float* g = A + (size_t)(rowBase + row) * Kdim + kb + c4;
      // flat LDS pointer: addr[31:0] is the LDS byte offset (ISA 10.2)
      unsigned lds = (unsigned)(size_t)&lAf[buf][row][c4];
      asm volatile("global_load_async_to_lds_b128 %0, %1, off"
                   :: "v"(lds), "v"(g) : "memory");
    }
  };
  // ---- sync stage of B tile (needs f32->f16 + transpose) ----
  auto stageB = [&](int buf, int kb) {
#pragma unroll
    for (int j = 0; j < 2; ++j) {
      int i  = tid + 256 * j;
      int kr = i >> 4;
      int c4 = (i & 15) * 4;
      float4 v = *(const float4*)(W + (size_t)(kb + kr) * Ncols + colBase + c4);
      lB[buf][c4 + 0][kr] = (_Float16)v.x;
      lB[buf][c4 + 1][kr] = (_Float16)v.y;
      lB[buf][c4 + 2][kr] = (_Float16)v.z;
      lB[buf][c4 + 3][kr] = (_Float16)v.w;
    }
  };

  stageA(0, 0);
  stageB(0, 0);
  asm volatile("s_wait_asynccnt 0x0" ::: "memory");
  __syncthreads();

  v8f c00 = {}; v8f c01 = {}; v8f c10 = {}; v8f c11 = {};

  int buf = 0;
  for (int kb = 0; kb < Kdim; kb += 32) {
    const int nkb = kb + 32;
    if (nkb < Kdim) {
      __builtin_prefetch(W + (size_t)nkb * Ncols + colBase, 0, 1);
      stageA(buf ^ 1, nkb);
      stageB(buf ^ 1, nkb);
    }

    // ---- A fragments (lanes 0-15: K 0-7/16-23; 16-31: K 8-15/24-31) ----
    const int ar = wr * 32 + laneLo;
    const int ak = laneHi * 8;
    v16h a0, a1;
    {
      const float* ap = &lAf[buf][ar][ak];
#pragma unroll
      for (int i = 0; i < 8; ++i) {
        a0[i] = (_Float16)ap[i];
        a0[i + 8] = (_Float16)ap[i + 16];
      }
      const float* ap2 = &lAf[buf][ar + 16][ak];
#pragma unroll
      for (int i = 0; i < 8; ++i) {
        a1[i] = (_Float16)ap2[i];
        a1[i + 8] = (_Float16)ap2[i + 16];
      }
    }
    // ---- B fragments (lane = column; lanes 0-15: K 0-7/16-23; 16-31: K 8-15/24-31) ----
    const int bc = wc * 32 + laneLo;
    const int bk = laneHi * 8;
    v16h b0, b1;
    {
      v8h lo = *(const v8h*)&lB[buf][bc][bk];
      v8h hi = *(const v8h*)&lB[buf][bc][bk + 16];
#pragma unroll
      for (int i = 0; i < 8; ++i) { b0[i] = lo[i]; b0[i + 8] = hi[i]; }
      v8h lo2 = *(const v8h*)&lB[buf][bc + 16][bk];
      v8h hi2 = *(const v8h*)&lB[buf][bc + 16][bk + 16];
#pragma unroll
      for (int i = 0; i < 8; ++i) { b1[i] = lo2[i]; b1[i + 8] = hi2[i]; }
    }

    c00 = __builtin_amdgcn_wmma_f32_16x16x32_f16(false, a0, false, b0, (short)0, c00, false, false);
    c01 = __builtin_amdgcn_wmma_f32_16x16x32_f16(false, a0, false, b1, (short)0, c01, false, false);
    c10 = __builtin_amdgcn_wmma_f32_16x16x32_f16(false, a1, false, b0, (short)0, c10, false, false);
    c11 = __builtin_amdgcn_wmma_f32_16x16x32_f16(false, a1, false, b1, (short)0, c11, false, false);

    asm volatile("s_wait_asynccnt 0x0" ::: "memory");
    __syncthreads();
    buf ^= 1;
  }

  // ---- epilogue: C/D layout -> row = m*16 + v + laneHi*8, col = n*16 + laneLo
#pragma unroll
  for (int m = 0; m < 2; ++m) {
#pragma unroll
    for (int n = 0; n < 2; ++n) {
      v8f cc = (m == 0) ? (n == 0 ? c00 : c01) : (n == 0 ? c10 : c11);
      int col = colBase + wc * 32 + n * 16 + laneLo;
      float bv = bias[col];
#pragma unroll
      for (int v = 0; v < 8; ++v) {
        int row = rowBase + wr * 32 + m * 16 + laneHi * 8 + v;
        float val = cc[v] + bv;
        if (relu) val = fmaxf(val, 0.0f);
        Out[(size_t)row * Ncols + col] = val;
      }
    }
  }
}

// ---------------------------------------------------------------------------
// Fused softmax + bilinear sampling + attention weighting.
// One block per query row; thread = (head m, channel d).
// values fits in L2 (44.6 MB < 192 MB) -> gather is L2-bound.
// ---------------------------------------------------------------------------
__global__ __launch_bounds__(256)
void sample_kernel(const float* __restrict__ values,
                   const float* __restrict__ offb,
                   const float* __restrict__ logits,
                   const float* __restrict__ ref0, const float* __restrict__ ref1,
                   const float* __restrict__ ref2, const float* __restrict__ ref3,
                   float* __restrict__ acc) {
  const int r  = blockIdx.x;
  const int m  = threadIdx.x >> 5;
  const int d  = threadIdx.x & 31;
  const int b  = r / QSUM;
  const int pq = r - b * QSUM;

  int p; const float* refp; int hwq;
  if (pq < 16384)      { p = pq;          refp = ref0; hwq = 16384; }
  else if (pq < 20480) { p = pq - 16384;  refp = ref1; hwq = 4096;  }
  else if (pq < 21504) { p = pq - 20480;  refp = ref2; hwq = 1024;  }
  else                 { p = pq - 21504;  refp = ref3; hwq = 256;   }
  const float rx = refp[(size_t)(b * hwq + p) * 2 + 0];
  const float ry = refp[(size_t)(b * hwq + p) * 2 + 1];

  // softmax over L*K = 16 logits of head m
  const float* lg = logits + (size_t)r * 128 + m * 16;
  float w[16];
  float mx = -3.4e38f;
#pragma unroll
  for (int j = 0; j < 16; ++j) mx = fmaxf(mx, lg[j]);
  float s = 0.0f;
#pragma unroll
  for (int j = 0; j < 16; ++j) { w[j] = __expf(lg[j] - mx); s += w[j]; }
  const float inv = 1.0f / s;

  const float* op = offb + (size_t)r * 256 + m * 32;   // (l,k,2) within head

  const int Ws[4]  = {128, 64, 32, 16};
  const int Hs[4]  = {128, 64, 32, 16};
  const int LO[4]  = {0, 16384, 20480, 21504};

  float sum = 0.0f;
#pragma unroll
  for (int l = 0; l < 4; ++l) {
    const int Wl = Ws[l], Hl = Hs[l];
    const int rowb = b * QSUM + LO[l];
#pragma unroll
    for (int k = 0; k < 4; ++k) {
      const float ox = op[(l * 4 + k) * 2 + 0];
      const float oy = op[(l * 4 + k) * 2 + 1];
      // (ref + off/scale) * scale - 0.5  ==  ref*scale + off - 0.5
      const float px = rx * (float)Wl + ox - 0.5f;
      const float py = ry * (float)Hl + oy - 0.5f;
      const float x0f = floorf(px), y0f = floorf(py);
      const float wx = px - x0f, wy = py - y0f;
      const int x0 = (int)x0f, y0 = (int)y0f;
      const float aw = w[l * 4 + k] * inv;
      float g = 0.0f;
#pragma unroll
      for (int dy = 0; dy < 2; ++dy) {
#pragma unroll
        for (int dx = 0; dx < 2; ++dx) {
          const int xi = x0 + dx, yi = y0 + dy;
          if (xi >= 0 && xi < Wl && yi >= 0 && yi < Hl) {
            const float cw = (dx ? wx : 1.0f - wx) * (dy ? wy : 1.0f - wy);
            const size_t vrow = (size_t)(rowb + yi * Wl + xi);
            g += cw * values[vrow * CCH + m * 32 + d];
          }
        }
      }
      sum += aw * g;
    }
  }
  acc[(size_t)r * CCH + m * 32 + d] = sum;
}

// ---------------------------------------------------------------------------
// out = LayerNorm(a + res) * g + beta    (one block of 256 per row, C = 256)
// ---------------------------------------------------------------------------
__global__ __launch_bounds__(256)
void addln_kernel(const float* __restrict__ a, const float* __restrict__ res,
                  const float* __restrict__ g, const float* __restrict__ beta,
                  float* __restrict__ out) {
  const int r = blockIdx.x;
  const int c = threadIdx.x;
  __shared__ float red[256];
  float v = a[(size_t)r * CCH + c] + res[(size_t)r * CCH + c];

  red[c] = v;
  __syncthreads();
  for (int s = 128; s > 0; s >>= 1) {
    if (c < s) red[c] += red[c + s];
    __syncthreads();
  }
  const float mu = red[0] * (1.0f / 256.0f);
  __syncthreads();

  const float dvv = v - mu;
  red[c] = dvv * dvv;
  __syncthreads();
  for (int s = 128; s > 0; s >>= 1) {
    if (c < s) red[c] += red[c + s];
    __syncthreads();
  }
  const float var = red[0] * (1.0f / 256.0f);
  out[(size_t)r * CCH + c] = dvv * rsqrtf(var + 1e-5f) * g[c] + beta[c];
}

// ---------------------------------------------------------------------------
extern "C" void kernel_launch(void* const* d_in, const int* in_sizes, int n_in,
                              void* d_out, int out_size, void* d_ws, size_t ws_size,
                              hipStream_t stream) {
  (void)in_sizes; (void)n_in; (void)out_size; (void)ws_size;

  const float* feat[4] = {(const float*)d_in[0], (const float*)d_in[3],
                          (const float*)d_in[6], (const float*)d_in[9]};
  const float* pos[4]  = {(const float*)d_in[1], (const float*)d_in[4],
                          (const float*)d_in[7], (const float*)d_in[10]};
  const float* ref[4]  = {(const float*)d_in[2], (const float*)d_in[5],
                          (const float*)d_in[8], (const float*)d_in[11]};
  const float* W_val  = (const float*)d_in[12];
  const float* b_val  = (const float*)d_in[13];
  const float* W_off  = (const float*)d_in[14];
  const float* b_off  = (const float*)d_in[15];
  const float* W_attn = (const float*)d_in[16];
  const float* b_attn = (const float*)d_in[17];
  const float* W_out  = (const float*)d_in[18];
  const float* b_out  = (const float*)d_in[19];
  const float* g1     = (const float*)d_in[20];
  const float* be1    = (const float*)d_in[21];
  const float* g2     = (const float*)d_in[22];
  const float* be2    = (const float*)d_in[23];
  const float* W1     = (const float*)d_in[24];
  const float* b1     = (const float*)d_in[25];
  const float* W2     = (const float*)d_in[26];
  const float* b2     = (const float*)d_in[27];

  const size_t NQ = NQ_TOTAL;
  float* ws      = (float*)d_ws;
  float* zf      = ws;
  float* values  = zf      + NQ * 256;
  float* offb    = values  + NQ * 256;
  float* logitsb = offb    + NQ * 256;
  float* accb    = logitsb + NQ * 128;
  float* attb    = accb    + NQ * 256;
  float* hb      = attb    + NQ * 256;
  float* xb      = offb;   // reuse: off is dead after sampling
  float* yb      = accb;   // reuse: acc is dead after attn-out GEMM

  // 1. zf = feat + pos (relayout to [b, level-concat, C])
  const int HW[4] = {16384, 4096, 1024, 256};
  const int LO[4] = {0, 16384, 20480, 21504};
  for (int l = 0; l < 4; ++l) {
    const int nvec = 2 * HW[l] * 64;   // B * HW * C/4 float4s
    addpos_kernel<<<nvec / 256, 256, 0, stream>>>(
        (const float4*)feat[l], (const float4*)pos[l], zf, HW[l], LO[l]);
  }

  const dim3 blk(256);
  auto grid = [](int ncols, int nrows) { return dim3(ncols / 64, nrows / 128); };

  // 2. projections (WMMA GEMMs)
  gemm_wmma_kernel<<<grid(256, NQ_TOTAL), blk, 0, stream>>>(zf, W_val, b_val, values,
                                                            NQ_TOTAL, 256, 256, 0);
  gemm_wmma_kernel<<<grid(256, NQ_TOTAL), blk, 0, stream>>>(zf, W_off, b_off, offb,
                                                            NQ_TOTAL, 256, 256, 0);
  gemm_wmma_kernel<<<grid(128, NQ_TOTAL), blk, 0, stream>>>(zf, W_attn, b_attn, logitsb,
                                                            NQ_TOTAL, 256, 128, 0);

  // 3. fused softmax + deformable bilinear sampling + attention weighting
  sample_kernel<<<NQ_TOTAL, blk, 0, stream>>>(values, offb, logitsb,
                                              ref[0], ref[1], ref[2], ref[3], accb);

  // 4. attention output projection + residual LN
  gemm_wmma_kernel<<<grid(256, NQ_TOTAL), blk, 0, stream>>>(accb, W_out, b_out, attb,
                                                            NQ_TOTAL, 256, 256, 0);
  addln_kernel<<<NQ_TOTAL, blk, 0, stream>>>(zf, attb, g1, be1, xb);

  // 5. FFN (dominant 91 GF) + residual LN -> d_out
  gemm_wmma_kernel<<<grid(2048, NQ_TOTAL), blk, 0, stream>>>(xb, W1, b1, hb,
                                                             NQ_TOTAL, 256, 2048, 1);
  gemm_wmma_kernel<<<grid(256, NQ_TOTAL), blk, 0, stream>>>(hb, W2, b2, yb,
                                                            NQ_TOTAL, 2048, 256, 0);
  addln_kernel<<<NQ_TOTAL, blk, 0, stream>>>(xb, yb, g2, be2, (float*)d_out);
}
